// SimplifiedGNN_52158082843042
// MI455X (gfx1250) — matrix operand order repaired and checked by reading, not compiled, not run on hardware
//
#include <hip/hip_runtime.h>

// ---------------------------------------------------------------------------
// CDNA5 (gfx1250) GCN forward: bf16 WMMA GEMMs + streamed f32 atomic scatter.
// Round 3: hoist full A-row load+convert out of the K loop so global loads
// issue as one clause and the WMMA loop is pure ds_load+wmma.
// ---------------------------------------------------------------------------

typedef __attribute__((ext_vector_type(16))) __bf16 v16bf;
typedef __attribute__((ext_vector_type(16))) float  v16f;
typedef __attribute__((ext_vector_type(8)))  float  v8f;

__device__ __forceinline__ __bf16 f2bf(float f) {
  return (__bf16)f;                              // native v_cvt to bf16
}

// ---- degree / norm ---------------------------------------------------------

__global__ void k_init_deg(float* __restrict__ deg, int n) {
  int i = blockIdx.x * blockDim.x + threadIdx.x;
  if (i < n) deg[i] = 1.0f;                      // self loop
}

__global__ void k_count_deg(const int* __restrict__ col, float* __restrict__ deg, int e) {
  int i = blockIdx.x * blockDim.x + threadIdx.x;
  if (i < e) unsafeAtomicAdd(&deg[col[i]], 1.0f);
}

__global__ void k_rsqrt(float* __restrict__ d, int n) {
  int i = blockIdx.x * blockDim.x + threadIdx.x;
  if (i < n) d[i] = rsqrtf(d[i]);
}

// ---- weight packing into WMMA B-fragment layout ---------------------------
// Fragment (kc, nt) = 32x16 bf16 sub-block of W (K = kc*32..+31, N = nt*16..+15),
// lane-major: lane l holds 16 K values of column (l & 15);
// lanes 0-15 -> K {0..7,16..23} of the chunk, lanes 16-31 -> K {8..15,24..31}.

__global__ void k_pack_w(const float* __restrict__ W, __bf16* __restrict__ out,
                         int K, int N) {
  int tid = blockIdx.x * blockDim.x + threadIdx.x;
  if (tid >= K * N) return;
  int frag = tid >> 9;             // 512 elements / fragment
  int lane = (tid >> 4) & 31;
  int j    = tid & 15;
  int ntiles = N >> 4;
  int kc = frag / ntiles;
  int nt = frag - kc * ntiles;
  int c  = nt * 16 + (lane & 15);
  int k  = kc * 32 + ((lane < 16) ? 0 : 8) + ((j < 8) ? j : (j + 8));
  out[tid] = f2bf(W[k * N + c]);
}

// ---- WMMA GEMM: out[M x (NT*16)] = (relu?)(H[M x 128]) @ W  (+bias) -------

template <int NT, bool RELU, bool BIAS>
__global__ void __launch_bounds__(256)
k_gemm(const float* __restrict__ H, const __bf16* __restrict__ Wp,
       const float* __restrict__ bias, float* __restrict__ out, int M) {
  constexpr int K = 128;
  constexpr int N = NT * 16;
  __shared__ unsigned int sW[NT * 1024];         // K*N bf16 = NT*1024 dwords

  // stage packed weights in LDS (shared by all 8 waves)
  {
    const unsigned int* src = (const unsigned int*)Wp;
    for (int i = threadIdx.x; i < NT * 1024; i += 256) sW[i] = src[i];
  }
  __syncthreads();

  const int lane = threadIdx.x & 31;
  const int wave = threadIdx.x >> 5;
  const int rowbase = blockIdx.x * 128 + wave * 16;

  int lrow = rowbase + (lane & 15);
  int r = (lrow < M) ? lrow : (M - 1);           // clamp loads, guard stores
  const float* __restrict__ hrow = H + (size_t)r * K;

  // --- Build all four A fragments up front (one load clause, then converts).
  // A fragment (16-bit A 16x32 layout): lanes 0-15 hold K {0..7,16..23},
  // lanes 16-31 hold K {8..15,24..31} of each 32-wide chunk.
  const int kshift = (lane < 16) ? 0 : 8;
  v16bf afrag[4];
#pragma unroll
  for (int kc = 0; kc < 4; ++kc) {
    int k0 = kc * 32 + kshift;
    float4 f0 = *(const float4*)(hrow + k0);
    float4 f1 = *(const float4*)(hrow + k0 + 4);
    float4 f2 = *(const float4*)(hrow + k0 + 16);
    float4 f3 = *(const float4*)(hrow + k0 + 20);
    v16f fa;
    fa[0]  = f0.x; fa[1]  = f0.y; fa[2]  = f0.z; fa[3]  = f0.w;
    fa[4]  = f1.x; fa[5]  = f1.y; fa[6]  = f1.z; fa[7]  = f1.w;
    fa[8]  = f2.x; fa[9]  = f2.y; fa[10] = f2.z; fa[11] = f2.w;
    fa[12] = f3.x; fa[13] = f3.y; fa[14] = f3.z; fa[15] = f3.w;
    if (RELU) {
      v16f zv = fa - fa;                         // zero vector
      fa = __builtin_elementwise_max(fa, zv);
    }
    afrag[kc] = __builtin_convertvector(fa, v16bf); // packed v_cvt_pk_bf16_f32
  }

  v8f acc[NT];
  v8f zero = {0.f, 0.f, 0.f, 0.f, 0.f, 0.f, 0.f, 0.f};
#pragma unroll
  for (int t = 0; t < NT; ++t) acc[t] = zero;

  const __bf16* sWb = (const __bf16*)sW;

  // --- Pure ds_load + wmma loop (no global traffic, no converts).
#pragma unroll
  for (int kc = 0; kc < 4; ++kc) {
#pragma unroll
    for (int nt = 0; nt < NT; ++nt) {
      const v16bf b = *(const v16bf*)(sWb + (((kc * NT + nt) << 9) + (lane << 4)));
      acc[nt] = __builtin_amdgcn_wmma_f32_16x16x32_bf16(
          /*neg_a=*/false, afrag[kc], /*neg_b=*/false, b,
          /*c_mod=*/(short)0, acc[nt], /*reuse_a=*/false, /*reuse_b=*/false);
    }
  }

  // C/D layout: VGPR v, lanes 0-15 -> (M=v, N=lane); lanes 16-31 -> (M=v+8, N=lane-16)
  const int m0   = rowbase + ((lane < 16) ? 0 : 8);
  const int ncol = lane & 15;
#pragma unroll
  for (int nt = 0; nt < NT; ++nt) {
    float badd = BIAS ? bias[nt * 16 + ncol] : 0.f;
#pragma unroll
    for (int v = 0; v < 8; ++v) {
      int rr = m0 + v;
      if (rr < M) out[(size_t)rr * N + nt * 16 + ncol] = acc[nt][v] + badd;
    }
  }
}

// ---- aggregation -----------------------------------------------------------

__global__ void k_agg_init(const float* __restrict__ t, const float* __restrict__ dinv,
                           const float* __restrict__ bias, float* __restrict__ agg, int n) {
  int i = blockIdx.x * blockDim.x + threadIdx.x;
  if (i >= n * 128) return;
  int node = i >> 7;
  int c = i & 127;
  float d = dinv[node];
  agg[i] = bias[c] + d * d * t[i];               // bias + self-loop term
}

__global__ void k_agg_edges(const int* __restrict__ rows, const int* __restrict__ cols,
                            const float* __restrict__ dinv, const float* __restrict__ t,
                            float* __restrict__ agg, int e) {
  int tid = blockIdx.x * blockDim.x + threadIdx.x;
  int ed = tid >> 2;
  if (ed >= e) return;
  int part = tid & 3;
  int r = rows[ed], c = cols[ed];
  float w = dinv[r] * dinv[c];
  const float4* __restrict__ src = (const float4*)(t + (size_t)r * 128 + part * 32);
  float* dst = agg + (size_t)c * 128 + part * 32;
#pragma unroll
  for (int q = 0; q < 8; ++q) {
    float4 v = src[q];
    unsafeAtomicAdd(dst + q * 4 + 0, w * v.x);
    unsafeAtomicAdd(dst + q * 4 + 1, w * v.y);
    unsafeAtomicAdd(dst + q * 4 + 2, w * v.z);
    unsafeAtomicAdd(dst + q * 4 + 3, w * v.w);
  }
}

// ---- global mean pool ------------------------------------------------------

__global__ void k_mean(const float* __restrict__ preds, float* __restrict__ out,
                       int n, int c) {
  __shared__ float s[256];
  int col = blockIdx.x;
  float sum = 0.f;
  for (int i = threadIdx.x; i < n; i += blockDim.x) sum += preds[(size_t)i * c + col];
  s[threadIdx.x] = sum;
  __syncthreads();
  for (int st = 128; st > 0; st >>= 1) {
    if ((int)threadIdx.x < st) s[threadIdx.x] += s[threadIdx.x + st];
    __syncthreads();
  }
  if (threadIdx.x == 0) out[col] = s[0] / (float)n;
}

// ---------------------------------------------------------------------------

extern "C" void kernel_launch(void* const* d_in, const int* in_sizes, int n_in,
                              void* d_out, int out_size, void* d_ws, size_t ws_size,
                              hipStream_t stream) {
  const float* x    = (const float*)d_in[0];
  const int*   ei   = (const int*)  d_in[1];
  const float* W1   = (const float*)d_in[2];
  const float* b1   = (const float*)d_in[3];
  const float* W2   = (const float*)d_in[4];
  const float* b2   = (const float*)d_in[5];
  const float* W3   = (const float*)d_in[6];
  const float* b3   = (const float*)d_in[7];
  const float* Wout = (const float*)d_in[8];
  const float* bout = (const float*)d_in[9];

  const int N = in_sizes[0] / 128;               // 50000 nodes
  const int E = in_sizes[1] / 2;                 // 800000 edges
  const int* rows = ei;                          // edge_index[0] = source
  const int* cols = ei + E;                      // edge_index[1] = target

  // workspace layout
  char* ws = (char*)d_ws;
  size_t off = 0;
  auto alloc = [&](size_t bytes) -> void* {
    void* p = ws + off;
    off = (off + bytes + 255) & ~(size_t)255;
    return p;
  };
  float*  dinv = (float*) alloc((size_t)N * 4);
  __bf16* W1p  = (__bf16*)alloc(128 * 128 * 2);
  __bf16* W2p  = (__bf16*)alloc(128 * 128 * 2);
  __bf16* W3p  = (__bf16*)alloc(128 * 128 * 2);
  __bf16* Wop  = (__bf16*)alloc(128 * 64 * 2);
  float*  bufA = (float*) alloc((size_t)N * 128 * 4);
  float*  bufB = (float*) alloc((size_t)N * 128 * 4);
  if (off > ws_size) return;                     // insufficient scratch

  auto cdiv = [](int a, int b) { return (a + b - 1) / b; };
  const int gNodes = cdiv(N, 256);
  const int gFeat  = cdiv(N * 128, 256);
  const int gEdge4 = cdiv(E * 4, 256);
  const int gGemm  = cdiv(N, 128);

  // 1) symmetric norm
  k_init_deg<<<gNodes, 256, 0, stream>>>(dinv, N);
  k_count_deg<<<cdiv(E, 256), 256, 0, stream>>>(cols, dinv, E);
  k_rsqrt<<<gNodes, 256, 0, stream>>>(dinv, N);

  // 2) pack weights to bf16 WMMA fragments
  k_pack_w<<<cdiv(128 * 128, 256), 256, 0, stream>>>(W1, W1p, 128, 128);
  k_pack_w<<<cdiv(128 * 128, 256), 256, 0, stream>>>(W2, W2p, 128, 128);
  k_pack_w<<<cdiv(128 * 128, 256), 256, 0, stream>>>(W3, W3p, 128, 128);
  k_pack_w<<<cdiv(128 * 64, 256), 256, 0, stream>>>(Wout, Wop, 128, 64);

  // 3) layer 1: x @ W1 -> bufA ; aggregate -> bufB
  k_gemm<8, false, false><<<gGemm, 256, 0, stream>>>(x, W1p, nullptr, bufA, N);
  k_agg_init<<<gFeat, 256, 0, stream>>>(bufA, dinv, b1, bufB, N);
  k_agg_edges<<<gEdge4, 256, 0, stream>>>(rows, cols, dinv, bufA, bufB, E);

  // 4) layer 2: relu(bufB) @ W2 -> bufA ; aggregate -> bufB
  k_gemm<8, true, false><<<gGemm, 256, 0, stream>>>(bufB, W2p, nullptr, bufA, N);
  k_agg_init<<<gFeat, 256, 0, stream>>>(bufA, dinv, b2, bufB, N);
  k_agg_edges<<<gEdge4, 256, 0, stream>>>(rows, cols, dinv, bufA, bufB, E);

  // 5) layer 3: relu(bufB) @ W3 -> bufA ; aggregate -> bufB
  k_gemm<8, true, false><<<gGemm, 256, 0, stream>>>(bufB, W3p, nullptr, bufA, N);
  k_agg_init<<<gFeat, 256, 0, stream>>>(bufA, dinv, b3, bufB, N);
  k_agg_edges<<<gEdge4, 256, 0, stream>>>(rows, cols, dinv, bufA, bufB, E);

  // 6) output projection: relu(bufB) @ Wout + bout -> d_out[0 : N*64]
  float* preds = (float*)d_out;
  k_gemm<4, true, true><<<gGemm, 256, 0, stream>>>(bufB, Wop, bout, preds, N);

  // 7) global mean pool -> d_out[N*64 : N*64+64]
  k_mean<<<64, 256, 0, stream>>>(preds, preds + (size_t)N * 64, N, 64);
}
